// Attention_20126216749365
// MI455X (gfx1250) — matrix operand order repaired and checked by reading, not compiled, not run on hardware
//
#include <hip/hip_runtime.h>
#include <hip/hip_bf16.h>

typedef __attribute__((ext_vector_type(16))) __bf16 v16bf;
typedef __attribute__((ext_vector_type(8)))  float  v8f;

constexpr int BATCH = 8;
constexpr int TQ    = 2048;
constexpr int TV    = 2048;
constexpr int DIM   = 256;

constexpr int BM    = 64;   // query rows per workgroup
constexpr int BN    = 64;   // kv rows per block (4 x 16 N-tiles)
constexpr int NT    = BN / 16;
constexpr int WAVES = 4;    // 16 query rows per wave
constexpr float NEGBIG = -1.0e30f;

__device__ __forceinline__ float rowmax16(float v) {
    v = fmaxf(v, __shfl_xor(v, 1, 32));
    v = fmaxf(v, __shfl_xor(v, 2, 32));
    v = fmaxf(v, __shfl_xor(v, 4, 32));
    v = fmaxf(v, __shfl_xor(v, 8, 32));
    return v;
}
__device__ __forceinline__ float rowsum16(float v) {
    v += __shfl_xor(v, 1, 32);
    v += __shfl_xor(v, 2, 32);
    v += __shfl_xor(v, 4, 32);
    v += __shfl_xor(v, 8, 32);
    return v;
}

// Async global->LDS copy of 16B per lane (CDNA5, ASYNCcnt-tracked).
__device__ __forceinline__ void async_copy_b128(uint32_t lds_byte_addr, const void* gaddr) {
    asm volatile("global_load_async_to_lds_b128 %0, %1, off"
                 :: "v"(lds_byte_addr), "v"(gaddr)
                 : "memory");
}
__device__ __forceinline__ void wait_asynccnt0() {
    asm volatile("s_wait_asynccnt 0" ::: "memory");
}

// ---------------------------------------------------------------------------
// Pre-pass: one block per (batch, kv-block of 64 rows).
//   Kws layout: [B][TV][DIM] bf16        (each 64x256 block = contiguous 32 KB)
//   Vws layout: [B*TV/BN][DIM][BN] bf16  (block-transposed, contiguous 32 KB)
// ---------------------------------------------------------------------------
__global__ __launch_bounds__(256)
void convert_kv_kernel(const float* __restrict__ K, const float* __restrict__ V,
                       __bf16* __restrict__ Kws, __bf16* __restrict__ Vws) {
    const int blk = blockIdx.x;               // b * (TV/BN) + kvb
    const int b   = blk / (TV / BN);
    const int kv0 = (blk % (TV / BN)) * BN;

    __shared__ __align__(16) __bf16 tile[DIM * BN];   // Vt staging [d][kv], 32 KB

    const float* Kb = K + ((size_t)b * TV + kv0) * DIM;
    const float* Vb = V + ((size_t)b * TV + kv0) * DIM;
    __bf16*      Ko = Kws + ((size_t)b * TV + kv0) * DIM;
    __bf16*      Vo = Vws + (size_t)blk * (DIM * BN);

    for (int base = threadIdx.x * 4; base < BN * DIM; base += 256 * 4) {
        const int r = base >> 8;              // kv row within block (0..63)
        const int c = base & (DIM - 1);       // d
        const float4 kf = *(const float4*)(Kb + base);
        Ko[base + 0] = (__bf16)kf.x;  Ko[base + 1] = (__bf16)kf.y;
        Ko[base + 2] = (__bf16)kf.z;  Ko[base + 3] = (__bf16)kf.w;
        const float4 vf = *(const float4*)(Vb + base);
        tile[(c + 0) * BN + r] = (__bf16)vf.x;
        tile[(c + 1) * BN + r] = (__bf16)vf.y;
        tile[(c + 2) * BN + r] = (__bf16)vf.z;
        tile[(c + 3) * BN + r] = (__bf16)vf.w;
    }
    __syncthreads();
    const uint4* src = (const uint4*)tile;
    uint4*       dst = (uint4*)Vo;
    for (int i = threadIdx.x; i < (DIM * BN * 2) / 16; i += 256) dst[i] = src[i];
}

// ---------------------------------------------------------------------------
// Flash attention (causal), double-buffered LDS pipeline.
// PRE=true : bf16 tiles DMA'd from d_ws with global_load_async_to_lds_b128,
//            next block's DMA overlaps current block's WMMA work.
// PRE=false: fallback converting fp32->bf16 in-kernel (same pipeline shape).
// ---------------------------------------------------------------------------
template <bool PRE>
__global__ __launch_bounds__(128, 1)
void fa_causal_kernel(const float* __restrict__ Q,
                      const float* __restrict__ V,
                      const float* __restrict__ K,
                      const __bf16* __restrict__ Kws,
                      const __bf16* __restrict__ Vws,
                      float* __restrict__ O) {
    __shared__ __align__(32) __bf16 sK [2][BN][DIM];     // 2 x 32 KB, row-major
    __shared__ __align__(32) __bf16 sVt[2][DIM][BN];     // 2 x 32 KB, transposed
    __shared__ __align__(32) __bf16 sP [WAVES][16][BN];  // 8 KB, per-wave P staging

    const int b    = blockIdx.y;
    const int qb   = blockIdx.x * BM;
    const int tid  = threadIdx.x;
    const int wave = tid >> 5;
    const int lane = tid & 31;
    const int half = lane >> 4;
    const int l16  = lane & 15;

    const float* Qb = Q + (size_t)b * TQ * DIM;
    const float* Kb = K + (size_t)b * TV * DIM;
    const float* Vb = V + (size_t)b * TV * DIM;
    float*       Ob = O + (size_t)b * TQ * DIM;

    // ---- Q tile for this wave: bf16 A-fragments ----------------------------
    const int qrow = qb + wave * 16 + l16;
    v16bf qa[8];
#pragma unroll
    for (int c = 0; c < 8; ++c) {
        const float* src = Qb + (size_t)qrow * DIM + c * 32 + 16 * half;
#pragma unroll
        for (int i = 0; i < 8; ++i) qa[c][i] = (__bf16)src[i];
#pragma unroll
        for (int i = 0; i < 8; ++i) qa[c][8 + i] = (__bf16)src[8 + i];
    }

    v8f o[16];
#pragma unroll
    for (int i = 0; i < 16; ++i)
        o[i] = (v8f){0.f, 0.f, 0.f, 0.f, 0.f, 0.f, 0.f, 0.f};
    float m[8], l[8];
#pragma unroll
    for (int r = 0; r < 8; ++r) { m[r] = NEGBIG; l[r] = 0.f; }

    const float scale = 0.0625f;             // 1/sqrt(256)
    const int nblocks = (qb + BM) / BN;      // causal bound, >= 1

    // per-buffer staging helper -------------------------------------------------
    auto stage = [&](int buf, int kv0) {
        if constexpr (PRE) {
            const char* Ksrc = (const char*)(Kws + ((size_t)b * TV + kv0) * DIM);
            const char* Vsrc = (const char*)(Vws +
                               ((size_t)(b * (TV / BN) + kv0 / BN)) * (DIM * BN));
            const uint32_t ldsK = (uint32_t)(uintptr_t)&sK [buf][0][0];
            const uint32_t ldsV = (uint32_t)(uintptr_t)&sVt[buf][0][0];
#pragma unroll
            for (int t = 0; t < (BN * DIM * 2) / (128 * 16); ++t) {   // 16 iters
                const uint32_t off = (uint32_t)(tid + t * 128) * 16;
                async_copy_b128(ldsK + off, Ksrc + off);
                async_copy_b128(ldsV + off, Vsrc + off);
            }
        } else {
            for (int base = tid * 4; base < BN * DIM; base += 128 * 4) {
                const int r = base >> 8;
                const int c = base & (DIM - 1);
                const float4 kf = *(const float4*)(Kb + (size_t)(kv0 + r) * DIM + c);
                sK[buf][r][c + 0] = (__bf16)kf.x;  sK[buf][r][c + 1] = (__bf16)kf.y;
                sK[buf][r][c + 2] = (__bf16)kf.z;  sK[buf][r][c + 3] = (__bf16)kf.w;
                const float4 vf = *(const float4*)(Vb + (size_t)(kv0 + r) * DIM + c);
                sVt[buf][c + 0][r] = (__bf16)vf.x; sVt[buf][c + 1][r] = (__bf16)vf.y;
                sVt[buf][c + 2][r] = (__bf16)vf.z; sVt[buf][c + 3][r] = (__bf16)vf.w;
            }
        }
    };

    // ---- prologue: fill buffer 0 with block 0 ------------------------------
    stage(0, 0);
    if constexpr (PRE) wait_asynccnt0();
    __syncthreads();

    for (int kb = 0; kb < nblocks; ++kb) {
        const int buf = kb & 1;
        const int kv0 = kb * BN;

        // kick off DMA for the next block into the other buffer (overlapped)
        if (kb + 1 < nblocks) stage(buf ^ 1, kv0 + BN);

        // ---- S = Q * K^T : four 16x16 N-tiles -------------------------------
        v8f s[NT];
#pragma unroll
        for (int t = 0; t < NT; ++t)
            s[t] = (v8f){0.f,0.f,0.f,0.f,0.f,0.f,0.f,0.f};
#pragma unroll
        for (int c = 0; c < 8; ++c) {
#pragma unroll
            for (int t = 0; t < NT; ++t) {
                const v16bf kb16 = *(const v16bf*)&sK[buf][t * 16 + l16][c * 32 + 16 * half];
                s[t] = __builtin_amdgcn_wmma_f32_16x16x32_bf16(false, qa[c], false, kb16,
                                                               (short)0, s[t], false, false);
            }
        }

        // ---- mask + online softmax -----------------------------------------
        float corrf[8];
#pragma unroll
        for (int r = 0; r < 8; ++r) {
            const int M  = r + 8 * half;
            const int qg = qb + wave * 16 + M;
            float vv[NT];
#pragma unroll
            for (int t = 0; t < NT; ++t) {
                vv[t] = s[t][r] * scale;
                if (kv0 + t * 16 + l16 > qg) vv[t] = NEGBIG;
            }
            float mx = vv[0];
#pragma unroll
            for (int t = 1; t < NT; ++t) mx = fmaxf(mx, vv[t]);
            mx = rowmax16(mx);
            const float mnew = fmaxf(m[r], mx);
            float p[NT], psum = 0.f;
#pragma unroll
            for (int t = 0; t < NT; ++t) { p[t] = __expf(vv[t] - mnew); psum += p[t]; }
            const float rs = rowsum16(psum);
            const float cf = __expf(m[r] - mnew);
            l[r] = l[r] * cf + rs;
            m[r] = mnew;
            corrf[r] = cf;
#pragma unroll
            for (int t = 0; t < NT; ++t) sP[wave][M][t * 16 + l16] = (__bf16)p[t];
        }

#pragma unroll
        for (int r = 0; r < 8; ++r) {
            const float cf = corrf[r];
#pragma unroll
            for (int dt = 0; dt < 16; ++dt) o[dt][r] *= cf;
        }

        // ---- reload P in A layout: two K-chunks of 32 -----------------------
        v16bf pa[2];
#pragma unroll
        for (int c = 0; c < 2; ++c) {
            const __bf16* pr = &sP[wave][l16][c * 32 + 16 * half];
#pragma unroll
            for (int i = 0; i < 8; ++i) pa[c][i]     = pr[i];
#pragma unroll
            for (int i = 0; i < 8; ++i) pa[c][8 + i] = pr[8 + i];
        }

        // ---- O += P * V -----------------------------------------------------
#pragma unroll
        for (int dt = 0; dt < 16; ++dt) {
#pragma unroll
            for (int c = 0; c < 2; ++c) {
                const v16bf vb = *(const v16bf*)&sVt[buf][dt * 16 + l16][c * 32 + 16 * half];
                o[dt] = __builtin_amdgcn_wmma_f32_16x16x32_bf16(false, pa[c], false, vb,
                                                                (short)0, o[dt], false, false);
            }
        }

        // drain next-block DMA, make all LDS writes visible for next iteration
        if constexpr (PRE) wait_asynccnt0();
        __syncthreads();
    }

    // ---- normalize and store ------------------------------------------------
#pragma unroll
    for (int r = 0; r < 8; ++r) {
        const int   M   = r + 8 * half;
        const float inv = 1.0f / l[r];
        float* dst = Ob + (size_t)(qb + wave * 16 + M) * DIM;
#pragma unroll
        for (int dt = 0; dt < 16; ++dt)
            dst[dt * 16 + l16] = o[dt][r] * inv;
    }
}

extern "C" void kernel_launch(void* const* d_in, const int* in_sizes, int n_in,
                              void* d_out, int out_size, void* d_ws, size_t ws_size,
                              hipStream_t stream) {
    // setup_inputs order: query, value, key, q_mask, v_mask
    const float* Q = (const float*)d_in[0];
    const float* V = (const float*)d_in[1];
    const float* K = (const float*)d_in[2];
    float*       O = (float*)d_out;

    const size_t need = (size_t)2 * BATCH * TV * DIM * sizeof(__bf16); // 16 MB
    dim3 grid(TQ / BM, BATCH);

    if (ws_size >= need) {
        __bf16* Kws = (__bf16*)d_ws;
        __bf16* Vws = Kws + (size_t)BATCH * TV * DIM;
        convert_kv_kernel<<<BATCH * (TV / BN), 256, 0, stream>>>(K, V, Kws, Vws);
        fa_causal_kernel<true><<<grid, 128, 0, stream>>>(Q, V, K, Kws, Vws, O);
    } else {
        fa_causal_kernel<false><<<grid, 128, 0, stream>>>(Q, V, K, nullptr, nullptr, O);
    }
}
